// EGNN_Sparse_Network_360777253438
// MI455X (gfx1250) — matrix operand order repaired and checked by reading, not compiled, not run on hardware
//
#include <hip/hip_runtime.h>
#include <cstdint>

typedef _Float16 f16;
typedef __attribute__((ext_vector_type(16))) _Float16 v16h;
typedef __attribute__((ext_vector_type(8)))  _Float16 v8h;
typedef __attribute__((ext_vector_type(8)))  float    v8f;

#define POS   3
#define FEATS 128
#define ROW   131          // POS + FEATS
#define MD    32
#define EIN   514          // 2*EDGE_IN (edge MLP hidden width)
#define K1P   288          // 257 padded to 9*32
#define N1T   33           // ceil(514/16) output tiles
#define K2P   544          // 514 padded to 17*32
#define ASTR  288          // LDS stride (halves) for edge A tile (loads hoisted once)
#define H1STR 552          // LDS stride for edge hidden tile (1104B: conflict-free)
#define MSTR  40           // LDS stride for m_ij tile
#define KN1   160          // node MLP K (128+32) = 5*32
#define ANSTR 168
#define NH    256          // node MLP hidden width
#define HNSTR 264
#define ETILE 32           // edges per block (2 row-tiles share every B-fragment)

static __device__ __forceinline__ float silu_f(float x) {
  return x / (1.0f + __expf(-x));
}

static __device__ __forceinline__ v8f wmma16(v16h a, v16h b, v8f c) {
  return __builtin_amdgcn_wmma_f32_16x16x32_f16(false, a, false, b, (short)0, c, false, false);
}

// A-fragment (16x32 f16) from row-major LDS tile.
// ISA layout: lanes 0-15 hold row M=lane, K {0..7, 16..23}; lanes 16-31 hold
// row M=lane-16, K {8..15, 24..31}.  -> two 16B LDS loads per lane.
static __device__ __forceinline__ v16h load_a_frag(const _Float16* base, int stride, int kk, int lane) {
  int row   = lane & 15;
  int kbase = kk * 32 + (lane >> 4) * 8;
  const _Float16* p = base + row * stride + kbase;
  v8h lo = *(const v8h*)p;
  v8h hi = *(const v8h*)(p + 16);
  v16h a;
#pragma unroll
  for (int t = 0; t < 8; ++t) { a[t] = lo[t]; a[t + 8] = hi[t]; }
  return a;
}

// B-fragment (32x16 f16) from pre-transposed global weights Bt[Npad][Kpad].
// ISA layout: lane&15 = column N, lanes 0-15 hold K {0..7, 16..23}, lanes 16-31
// hold K {8..15, 24..31} -> two 16B loads per lane.
static __device__ __forceinline__ v16h load_b_frag(const _Float16* __restrict__ B, int kpad, int ntile, int kk, int lane) {
  int n     = ntile * 16 + (lane & 15);
  int kbase = kk * 32 + (lane >> 4) * 8;
  const _Float16* p = B + (size_t)n * kpad + kbase;
  v8h lo = *(const v8h*)p;
  v8h hi = *(const v8h*)(p + 16);
  v16h b;
#pragma unroll
  for (int t = 0; t < 8; ++t) { b[t] = lo[t]; b[t + 8] = hi[t]; }
  return b;
}

// ---------------- weight transpose + pad + f32->f16 ----------------
__global__ void convT_pad_kernel(const float* __restrict__ src, _Float16* __restrict__ dst,
                                 int K, int N, int Kpad, int Npad) {
  int idx = blockIdx.x * blockDim.x + threadIdx.x;
  int total = Kpad * Npad;
  if (idx >= total) return;
  int n = idx / Kpad;
  int k = idx - n * Kpad;
  float v = (n < N && k < K) ? src[(size_t)k * N + n] : 0.0f;
  dst[(size_t)n * Kpad + k] = (_Float16)v;
}

__global__ void zero_f32_kernel(float* __restrict__ p, long n) {
  long i = (long)blockIdx.x * blockDim.x + threadIdx.x;
  if (i < n) p[i] = 0.0f;
}

// ---------------- edge kernel: one 32-edge tile per 4-wave block ----------------
__global__ __launch_bounds__(128) void egnn_edge_kernel(
    const float* __restrict__ x, const int* __restrict__ eidx, int E,
    const _Float16* __restrict__ W1t, const float* __restrict__ b1,
    const _Float16* __restrict__ W2t, const float* __restrict__ b2,
    const _Float16* __restrict__ cW1t, const float* __restrict__ cb1,
    const float* __restrict__ cW2, const float* __restrict__ cb2,
    float* __restrict__ m_acc, float* __restrict__ w_acc, float* __restrict__ r_acc)
{
  __shared__ _Float16 A[ETILE * ASTR];     // [feats_i | feats_j | dist | 0-pad] f16
  __shared__ _Float16 H1[ETILE * H1STR];   // silu(h@W1+b1) f16, K2-padded
  __shared__ _Float16 Mh[ETILE * MSTR];    // m_ij f16
  __shared__ float    sRel[ETILE][3];
  __shared__ int      sIdx[ETILE];
  __shared__ float    sW[ETILE];

  const int tid  = threadIdx.x;
  const int lane = tid & 31;
  const int wave = tid >> 5;
  const int e0   = blockIdx.x * ETILE;

  { // gather phase: 4 threads per edge
    int e    = tid >> 2;
    int part = tid & 3;
    int eg   = e0 + e; if (eg >= E) eg = E - 1;
    int jn   = eidx[eg];
    int in_  = eidx[E + eg];
    const float* fi = x + (size_t)in_ * ROW + POS;
    const float* fj = x + (size_t)jn  * ROW + POS;
    _Float16* Ar = A + e * ASTR;
#pragma unroll
    for (int t = 0; t < 32; ++t) {
      int k = part * 32 + t;
      Ar[k]         = (_Float16)fi[k];
      Ar[FEATS + k] = (_Float16)fj[k];
    }
    if (part == 0) {
      float rx = x[(size_t)jn * ROW + 0] - x[(size_t)in_ * ROW + 0];
      float ry = x[(size_t)jn * ROW + 1] - x[(size_t)in_ * ROW + 1];
      float rz = x[(size_t)jn * ROW + 2] - x[(size_t)in_ * ROW + 2];
      float d  = sqrtf(rx * rx + ry * ry + rz * rz);
      Ar[256] = (_Float16)d;
      sRel[e][0] = rx; sRel[e][1] = ry; sRel[e][2] = rz;
      sIdx[e] = in_;
      sW[e]   = cb2[0];           // init edge-gate accumulator
    }
    if (part == 1) {
      for (int k = 257; k < K1P; ++k) Ar[k] = (_Float16)0.0f;
    }
  }
  // zero H1 pad columns 514..543 (read by the K=544 loop of GEMM2)
  for (int z = tid; z < ETILE * 30; z += 128) {
    int r = z / 30, c = EIN + (z - r * 30);
    H1[r * H1STR + c] = (_Float16)0.0f;
  }
  __syncthreads();

  // ---- GEMM1: (32 x 288) @ (288 x 514) -> silu -> H1 ----
  // Two 16-row A tiles share every B fragment: halves L2 B-traffic per FLOP.
  for (int nt = wave; nt < N1T; nt += 4) {
    int ntn = (nt + 4 < N1T) ? nt + 4 : nt;
    __builtin_prefetch(W1t + (size_t)ntn * 16 * K1P, 0, 3);  // global_prefetch_b8
    v8f acc0 = {};
    v8f acc1 = {};
#pragma unroll
    for (int kk = 0; kk < 9; ++kk) {
      v16h b  = load_b_frag(W1t, K1P, nt, kk, lane);
      v16h a0 = load_a_frag(A,             ASTR, kk, lane);
      v16h a1 = load_a_frag(A + 16 * ASTR, ASTR, kk, lane);
      acc0 = wmma16(a0, b, acc0);
      acc1 = wmma16(a1, b, acc1);
    }
    int ncol   = nt * 16 + (lane & 15);
    float bias = (ncol < EIN) ? b1[ncol] : 0.0f;
#pragma unroll
    for (int r = 0; r < 8; ++r) {
      int row = r + 8 * (lane >> 4);
      H1[row * H1STR + ncol]        = (_Float16)silu_f(acc0[r] + bias);
      H1[(row + 16) * H1STR + ncol] = (_Float16)silu_f(acc1[r] + bias);
    }
  }
  __syncthreads();

  // ---- GEMM2: (32 x 544) @ (544 x 32) -> silu -> m_ij; segment-sum m_i ----
  // 4 waves x one 16x16 tile each: rt = row tile, ct = col tile.
  {
    int rt = wave >> 1;
    int ct = wave & 1;
    v8f acc = {};
#pragma unroll
    for (int kk = 0; kk < 17; ++kk) {
      v16h a = load_a_frag(H1 + rt * 16 * H1STR, H1STR, kk, lane);
      v16h b = load_b_frag(W2t, K2P, ct, kk, lane);
      acc = wmma16(a, b, acc);
    }
    int ncol   = ct * 16 + (lane & 15);
    float bias = b2[ncol];
#pragma unroll
    for (int r = 0; r < 8; ++r) {
      int row = rt * 16 + r + 8 * (lane >> 4);
      float mv = silu_f(acc[r] + bias);
      Mh[row * MSTR + ncol] = (_Float16)mv;
      if (e0 + row < E) atomicAdd(&m_acc[(size_t)sIdx[row] * MD + ncol], mv);
    }
  }
  __syncthreads();

  // ---- GEMM3: (32 x 32) @ (32 x 128) -> silu -> dot with cW2 (shuffle-reduce) ----
  for (int t = wave; t < 16; t += 4) {
    int rt = t & 1;        // row tile (edges 0-15 / 16-31)
    int nt = t >> 1;       // col tile (0..7)
    v16h a = load_a_frag(Mh + rt * 16 * MSTR, MSTR, 0, lane);
    v16h b = load_b_frag(cW1t, 32, nt, 0, lane);
    v8f z = {};
    v8f acc = wmma16(a, b, z);
    int ncol   = nt * 16 + (lane & 15);
    float bias = cb1[ncol];
    float w2   = cW2[ncol];
#pragma unroll
    for (int r = 0; r < 8; ++r) {
      float p = silu_f(acc[r] + bias) * w2;
      // reduce across the 16 lanes that hold this row's columns
#pragma unroll
      for (int m = 1; m < 16; m <<= 1) p += __shfl_xor(p, m, 32);
      if ((lane & 15) == 0) {
        int row = rt * 16 + r + 8 * (lane >> 4);
        atomicAdd(&sW[row], p);   // ds_add_f32
      }
    }
  }
  __syncthreads();

  // ---- segment-sum w_i and r_i ----
  if (tid < ETILE) {
    int e = tid;
    if (e0 + e < E) {
      int idx = sIdx[e];
      atomicAdd(&w_acc[idx], sW[e]);
      atomicAdd(&r_acc[(size_t)idx * 3 + 0], sRel[e][0]);
      atomicAdd(&r_acc[(size_t)idx * 3 + 1], sRel[e][1]);
      atomicAdd(&r_acc[(size_t)idx * 3 + 2], sRel[e][2]);
    }
  }
}

// ---------------- node kernel: one 16-node tile per 4-wave block ----------------
__global__ __launch_bounds__(128) void egnn_node_kernel(
    const float* __restrict__ xin, float* __restrict__ xout, int N,
    const float* __restrict__ m_acc, const float* __restrict__ w_acc, const float* __restrict__ r_acc,
    const _Float16* __restrict__ nW1t, const float* __restrict__ nb1,
    const _Float16* __restrict__ nW2t, const float* __restrict__ nb2,
    const float* __restrict__ scale)
{
  __shared__ _Float16 An[16 * ANSTR];   // [feats | m_i] f16
  __shared__ _Float16 Hn[16 * HNSTR];   // silu(nh@nW1+nb1) f16

  const int tid  = threadIdx.x;
  const int lane = tid & 31;
  const int wave = tid >> 5;
  const int n0   = blockIdx.x * 16;

  { // gather [feats(128) | m_i(32)] as f16
    int r    = tid >> 3;
    int part = tid & 7;
    int n    = n0 + r; if (n >= N) n = N - 1;
    const float* f  = xin  + (size_t)n * ROW + POS;
    const float* mi = m_acc + (size_t)n * MD;
    _Float16* Ar = An + r * ANSTR;
#pragma unroll
    for (int t = 0; t < 20; ++t) {
      int k = part * 20 + t;
      float v = (k < FEATS) ? f[k] : mi[k - FEATS];
      Ar[k] = (_Float16)v;
    }
  }
  // coordinate update (tanh-gated normalized sum of rel vectors)
  if (tid < 16) {
    int n = n0 + tid;
    if (n < N) {
      float wv = tanhf(w_acc[n]);
      float rx = r_acc[(size_t)n * 3 + 0];
      float ry = r_acc[(size_t)n * 3 + 1];
      float rz = r_acc[(size_t)n * 3 + 2];
      float nrm = sqrtf(rx * rx + ry * ry + rz * rz);
      nrm = fmaxf(nrm, 1e-12f);
      float sc = wv * scale[0] / nrm;
      xout[(size_t)n * ROW + 0] = xin[(size_t)n * ROW + 0] + sc * rx;
      xout[(size_t)n * ROW + 1] = xin[(size_t)n * ROW + 1] + sc * ry;
      xout[(size_t)n * ROW + 2] = xin[(size_t)n * ROW + 2] + sc * rz;
    }
  }
  __syncthreads();

  // ---- GEMM n1: (16 x 160) @ (160 x 256) -> silu -> Hn ----
  for (int nt = wave; nt < 16; nt += 4) {
    v8f acc = {};
#pragma unroll
    for (int kk = 0; kk < 5; ++kk) {
      v16h a = load_a_frag(An, ANSTR, kk, lane);
      v16h b = load_b_frag(nW1t, KN1, nt, kk, lane);
      acc = wmma16(a, b, acc);
    }
    int ncol   = nt * 16 + (lane & 15);
    float bias = nb1[ncol];
#pragma unroll
    for (int r = 0; r < 8; ++r) {
      int row = r + 8 * (lane >> 4);
      Hn[row * HNSTR + ncol] = (_Float16)silu_f(acc[r] + bias);
    }
  }
  __syncthreads();

  // ---- GEMM n2: (16 x 256) @ (256 x 128) + bias + residual ----
  for (int nt = wave; nt < 8; nt += 4) {
    v8f acc = {};
#pragma unroll
    for (int kk = 0; kk < 8; ++kk) {
      v16h a = load_a_frag(Hn, HNSTR, kk, lane);
      v16h b = load_b_frag(nW2t, NH, nt, kk, lane);
      acc = wmma16(a, b, acc);
    }
    int ncol   = nt * 16 + (lane & 15);
    float bias = nb2[ncol];
#pragma unroll
    for (int r = 0; r < 8; ++r) {
      int row = r + 8 * (lane >> 4);
      int n   = n0 + row;
      if (n < N) {
        float res = xin[(size_t)n * ROW + POS + ncol];
        xout[(size_t)n * ROW + POS + ncol] = acc[r] + bias + res;
      }
    }
  }
}

// ---------------- host launcher ----------------
extern "C" void kernel_launch(void* const* d_in, const int* in_sizes, int n_in,
                              void* d_out, int out_size, void* d_ws, size_t ws_size,
                              hipStream_t stream)
{
  const float* x0   = (const float*)d_in[0];
  const int*   eidx = (const int*)d_in[1];
  const int N = in_sizes[0] / ROW;
  const int E = in_sizes[1] / 2;

  char* ws = (char*)d_ws;
  size_t off = 0;
  auto take = [&](size_t bytes) -> void* {
    void* p = ws + off;
    off = (off + bytes + 255) & ~(size_t)255;
    return p;
  };

  float* xb0   = (float*)take((size_t)N * ROW * sizeof(float));
  float* xb1   = (float*)take((size_t)N * ROW * sizeof(float));
  float* accum = (float*)take((size_t)N * 36 * sizeof(float)); // m_i(32) | w_i(1) | r_i(3)
  float* m_acc = accum;
  float* w_acc = accum + (size_t)N * 32;
  float* r_acc = accum + (size_t)N * 33;

  _Float16 *W1t[4], *W2t[4], *cW1t[4], *nW1t[4], *nW2t[4];
  for (int l = 0; l < 4; ++l) {
    W1t[l]  = (_Float16*)take((size_t)544 * K1P * sizeof(_Float16));
    W2t[l]  = (_Float16*)take((size_t)32  * K2P * sizeof(_Float16));
    cW1t[l] = (_Float16*)take((size_t)128 * 32  * sizeof(_Float16));
    nW1t[l] = (_Float16*)take((size_t)256 * KN1 * sizeof(_Float16));
    nW2t[l] = (_Float16*)take((size_t)128 * NH  * sizeof(_Float16));
  }

  // params layout: d_in[4 + l*13 + k], k in insertion order:
  // e_w1,e_b1,e_w2,e_b2,c_w1,c_b1,c_w2,c_b2,n_w1,n_b1,n_w2,n_b2,scale
  for (int l = 0; l < 4; ++l) {
    const float* ew1 = (const float*)d_in[4 + l * 13 + 0];
    const float* ew2 = (const float*)d_in[4 + l * 13 + 2];
    const float* cw1 = (const float*)d_in[4 + l * 13 + 4];
    const float* nw1 = (const float*)d_in[4 + l * 13 + 8];
    const float* nw2 = (const float*)d_in[4 + l * 13 + 10];
    auto conv = [&](const float* s, _Float16* d, int K, int Nn, int Kp, int Np) {
      int total = Kp * Np;
      convT_pad_kernel<<<(total + 255) / 256, 256, 0, stream>>>(s, d, K, Nn, Kp, Np);
    };
    conv(ew1, W1t[l],  257, 514, K1P, 544);
    conv(ew2, W2t[l],  514, 32,  K2P, 32);
    conv(cw1, cW1t[l], 32,  128, 32,  128);
    conv(nw1, nW1t[l], 160, 256, KN1, 256);
    conv(nw2, nW2t[l], 256, 128, NH,  128);
  }

  const float* xin = x0;
  for (int l = 0; l < 4; ++l) {
    float* xout = (l == 3) ? (float*)d_out : ((l & 1) ? xb1 : xb0);
    const float* b1  = (const float*)d_in[4 + l * 13 + 1];
    const float* b2  = (const float*)d_in[4 + l * 13 + 3];
    const float* cb1 = (const float*)d_in[4 + l * 13 + 5];
    const float* cw2 = (const float*)d_in[4 + l * 13 + 6];
    const float* cb2 = (const float*)d_in[4 + l * 13 + 7];
    const float* nb1 = (const float*)d_in[4 + l * 13 + 9];
    const float* nb2 = (const float*)d_in[4 + l * 13 + 11];
    const float* scl = (const float*)d_in[4 + l * 13 + 12];

    long zn = (long)N * 36;
    zero_f32_kernel<<<(int)((zn + 255) / 256), 256, 0, stream>>>(accum, zn);

    egnn_edge_kernel<<<(E + ETILE - 1) / ETILE, 128, 0, stream>>>(
        xin, eidx, E, W1t[l], b1, W2t[l], b2, cW1t[l], cb1, cw2, cb2,
        m_acc, w_acc, r_acc);

    egnn_node_kernel<<<(N + 15) / 16, 128, 0, stream>>>(
        xin, xout, N, m_acc, w_acc, r_acc, nW1t[l], nb1, nW2t[l], nb2, scl);

    xin = xout;
  }
  (void)ws_size; (void)n_in; (void)out_size;
}
